// FusionLayer_82583631167722
// MI455X (gfx1250) — matrix operand order repaired and checked by reading, not compiled, not run on hardware
//
#include <hip/hip_runtime.h>
#include <hip/hip_bf16.h>

// ---------------------------------------------------------------------------
// FusionLayer on MI455X (gfx1250, wave32, WMMA f32_16x16x32_f16)
// Fully fused: feature gather -> GEMM1 -> GEMM2 -> layer3 dot, all in LDS.
// 64-row blocks: each B fragment is reused by 4 M-tiles (4x less L2 traffic).
// ---------------------------------------------------------------------------
typedef __attribute__((ext_vector_type(16))) _Float16 v16h;
typedef __attribute__((ext_vector_type(8)))  _Float16 v8h;
typedef __attribute__((ext_vector_type(8)))  float    v8f;

#define RH   64
#define RW   64
#define RC   32
#define HF   96
#define WF   320
#define FEAT 256
#define NIN  518           // (256+3)*2
#define KPAD 544           // 17 * 32
#define N1   128
#define N2   64
#define NROWS (RH*RW*RC)   // 131072

#define MROWS 64           // rows per block (4 M-tiles of 16)
#define LDH1 136           // h1 tile LDS stride (halves), 272B row, 16B aligned
#define LDH2 72            // h2 tile LDS stride (halves), 144B row, 16B aligned

// packed weight sizes (halves)
#define B1_HALVES (17*8*32*16)   // 69632
#define B2_HALVES (4*4*32*16)    // 8192

// ---------------------------------------------------------------------------
// Pack w1/w2 into WMMA-B fragment-linear layout; compute M = P3 @ R0 @ Tr.
// B frag layout (dense 32x16 f16 B): lanes 0-15 hold column n=lane with
// K = ks*32 + 0..15; lanes 16-31 hold the same columns with K = ks*32+16..31.
// Packed so each lane's 16 halves are contiguous (one 32B load).
// ---------------------------------------------------------------------------
__global__ void pack_kernel(const float* __restrict__ w1,
                            const float* __restrict__ w2,
                            const float* __restrict__ Tr,
                            const float* __restrict__ R0,
                            const float* __restrict__ P3,
                            _Float16* __restrict__ B1,
                            _Float16* __restrict__ B2,
                            float* __restrict__ Mout) {
    int idx = blockIdx.x * blockDim.x + threadIdx.x;
    if (idx < B1_HALVES) {
        int q    = idx & 15;
        int lane = (idx >> 4) & 31;
        int t    = (idx >> 9) & 7;
        int s    = idx >> 12;
        int k    = s * 32 + (lane >> 4) * 16 + q;
        int n    = t * 16 + (lane & 15);
        float v  = (k < NIN) ? w1[k * N1 + n] : 0.0f;
        B1[idx] = (_Float16)v;
    } else if (idx < B1_HALVES + B2_HALVES) {
        int j    = idx - B1_HALVES;
        int q    = j & 15;
        int lane = (j >> 4) & 31;
        int t    = (j >> 9) & 3;
        int s    = j >> 11;
        int k    = s * 32 + (lane >> 4) * 16 + q;
        int n    = t * 16 + (lane & 15);
        B2[j] = (_Float16)w2[k * N2 + n];
    } else if (idx < B1_HALVES + B2_HALVES + 12) {
        int j = idx - B1_HALVES - B2_HALVES;
        int e = j >> 2, d = j & 3;
        float acc = 0.f;
        for (int i = 0; i < 4; ++i)
            for (int jj = 0; jj < 4; ++jj)
                acc += P3[e * 4 + i] * R0[i * 4 + jj] * Tr[jj * 4 + d];
        Mout[j] = acc;
    }
}

// ---------------------------------------------------------------------------
// Fully fused MLP over a 64-row tile.
// ---------------------------------------------------------------------------
__global__ __launch_bounds__(256) void fused_kernel(
    const float*    __restrict__ rgb,     // [HF*WF*FEAT]
    const float*    __restrict__ kdtree,  // [256*256*32*2*4]
    const float*    __restrict__ Mmat,    // [12]
    const _Float16* __restrict__ B1,      // packed w1
    const float*    __restrict__ b1,      // [128]
    const _Float16* __restrict__ B2,      // packed w2
    const float*    __restrict__ b2,      // [64]
    const float*    __restrict__ w3,      // [64]
    const float*    __restrict__ b3,      // [1]
    float*          __restrict__ mlp)     // [NROWS] per-row MLP output
{
    __shared__ _Float16 feats[MROWS * KPAD];    // 69632 B
    __shared__ _Float16 h1t[MROWS * LDH1];      // 17408 B
    __shared__ _Float16 h2t[MROWS * LDH2];      //  9216 B
    __shared__ int offs[2 * MROWS];             //   512 B

    const int tid = threadIdx.x;
    const int rowbase = blockIdx.x * MROWS;

    // ---- stage A: 128 leader lanes (one per (row, k)) ----------------------
    if (tid < 2 * MROWS) {
        int lr = tid >> 1;              // local row 0..63
        int k  = tid & 1;
        int r  = rowbase + lr;
        int h  = r >> 11;
        int w  = (r >> 5) & 63;
        int c  = r & 31;
        const float* kp = kdtree + (size_t)((((h * 4) * 256 + w * 4) * 32 + c) * 2 + k) * 4;
        float x0 = kp[0], x1 = kp[1], x2 = kp[2];
        int cb = k * 259 + 256;
        feats[lr * KPAD + cb + 0] = (_Float16)(x0 - 1.0f);
        feats[lr * KPAD + cb + 1] = (_Float16)(x1 - (float)(h * 4));
        feats[lr * KPAD + cb + 2] = (_Float16)(x2 - (float)(w * 4));
        float cam0 = Mmat[0] * x0 + Mmat[1] * x1 + Mmat[2]  * x2 + Mmat[3];
        float cam1 = Mmat[4] * x0 + Mmat[5] * x1 + Mmat[6]  * x2 + Mmat[7];
        float cam2 = Mmat[8] * x0 + Mmat[9] * x1 + Mmat[10] * x2 + Mmat[11];
        float z  = (fabsf(cam2) < 1e-6f) ? 1e-6f : cam2;
        int ty = (int)(cam1 / z);           // trunc toward zero (astype)
        int tx = (int)(cam0 / z);
        int iy = ty >> 2;                   // arithmetic shift == floor div 4
        int ix = tx >> 2;
        iy = iy < 0 ? 0 : (iy > HF - 1 ? HF - 1 : iy);
        ix = ix < 0 ? 0 : (ix > WF - 1 ? WF - 1 : ix);
        offs[tid] = (iy * WF + ix) * FEAT;
    }
    // zero-pad columns 518..543
    for (int i = tid; i < MROWS * 26; i += 256) {
        int lr = i / 26, cc = NIN + (i % 26);
        feats[lr * KPAD + cc] = (_Float16)0.0f;
    }
    __syncthreads();

    // ---- stage B: gather 256 rgb floats per (row,k); 2 threads per task ----
    {
        int task = tid >> 1;            // 0..127
        int sub  = tid & 1;             // 0..1 -> 128 floats each
        int lr   = task >> 1;
        int k    = task & 1;
        const float* src = rgb + offs[task] + sub * 128;
        _Float16* dst = &feats[lr * KPAD + k * 259 + sub * 128];
        #pragma unroll
        for (int i = 0; i < 128; i += 4) {
            float4 v = *(const float4*)(src + i);
            dst[i + 0] = (_Float16)v.x;
            dst[i + 1] = (_Float16)v.y;
            dst[i + 2] = (_Float16)v.z;
            dst[i + 3] = (_Float16)v.w;
        }
    }
    __syncthreads();

    // ---- stage C: GEMM1 (K=544, 17 steps x 4 M-tiles), ReLU -> h1t ---------
    const int lane = tid & 31;
    const int wave = tid >> 5;          // n-tile 0..7
    const int m    = lane & 15;
    const int hi   = lane >> 4;
    {
        v8f acc0 = {}, acc1 = {}, acc2 = {}, acc3 = {};
        #pragma unroll
        for (int s = 0; s < 17; ++s) {
            // one B fragment, reused by 4 M-tiles
            v16h b = *(const v16h*)(B1 + ((size_t)(s * 8 + wave) * 32 + lane) * 16);
            // A frag (16-bit A 16x32; contiguous k order, same as B packing):
            //   lanes 0-15  hold K = {k0..k0+15}    of row M=lane
            //   lanes 16-31 hold K = {k0+16..k0+31} of row M=lane-16
            int ko = s * 32 + hi * 16;
            #pragma unroll
            for (int mt = 0; mt < 4; ++mt) {
                const _Float16* arow = &feats[(mt * 16 + m) * KPAD + ko];
                v8h lo = *(const v8h*)(arow);
                v8h hv = *(const v8h*)(arow + 8);
                v16h a;
                #pragma unroll
                for (int i = 0; i < 8; ++i) { a[i] = lo[i]; a[i + 8] = hv[i]; }
                v8f& acc = (mt == 0) ? acc0 : (mt == 1) ? acc1 : (mt == 2) ? acc2 : acc3;
                acc = __builtin_amdgcn_wmma_f32_16x16x32_f16(false, a, false, b,
                                                             (short)0, acc, false, false);
            }
        }
        // C/D layout: lane holds N = n0+(lane&15); M = j + 8*(lane>=16)
        int n = wave * 16 + (lane & 15);
        float bias = b1[n];
        int mb = hi * 8;
        #pragma unroll
        for (int mt = 0; mt < 4; ++mt) {
            v8f& acc = (mt == 0) ? acc0 : (mt == 1) ? acc1 : (mt == 2) ? acc2 : acc3;
            #pragma unroll
            for (int j = 0; j < 8; ++j) {
                float v = acc[j] + bias;
                v = v > 0.f ? v : 0.f;
                h1t[(mt * 16 + mb + j) * LDH1 + n] = (_Float16)v;
            }
        }
    }
    __syncthreads();

    // ---- stage D: GEMM2 (K=128, 4 steps), 16 wave-tiles over 8 waves -------
    {
        const int t = wave & 3;             // n-tile 0..3
        int n = t * 16 + (lane & 15);
        float bias = b2[n];
        int mb = hi * 8;
        #pragma unroll
        for (int iter = 0; iter < 2; ++iter) {
            int rt = (wave >> 2) + 2 * iter;    // row-tile 0..3
            const _Float16* arow = &h1t[(rt * 16 + m) * LDH1];
            v8f acc = {};
            #pragma unroll
            for (int s = 0; s < 4; ++s) {
                int ko = s * 32 + hi * 16;
                v8h lo = *(const v8h*)(arow + ko);
                v8h hv = *(const v8h*)(arow + ko + 8);
                v16h a;
                #pragma unroll
                for (int i = 0; i < 8; ++i) { a[i] = lo[i]; a[i + 8] = hv[i]; }
                v16h b = *(const v16h*)(B2 + ((size_t)(s * 4 + t) * 32 + lane) * 16);
                acc = __builtin_amdgcn_wmma_f32_16x16x32_f16(false, a, false, b,
                                                             (short)0, acc, false, false);
            }
            #pragma unroll
            for (int j = 0; j < 8; ++j) {
                float v = acc[j] + bias;
                v = v > 0.f ? v : 0.f;
                h2t[(rt * 16 + mb + j) * LDH2 + n] = (_Float16)v;
            }
        }
    }
    __syncthreads();

    // ---- stage E: layer3 64-dot per row -> one f32 per row -----------------
    if (tid < MROWS) {
        const _Float16* hr = &h2t[tid * LDH2];
        float d = b3[0];
        #pragma unroll
        for (int j = 0; j < N2; ++j) d += (float)hr[j] * w3[j];
        mlp[rowbase + tid] = d;
    }
}

// ---------------------------------------------------------------------------
// Transpose-reinterpret + 1x1 conv over C.
// y[p] (p = h'*64+w') = conv_b + sum_c conv_w[c] * mlp_row(m), where
// n = p*32 + c is decomposed as n = ci*4096 + h*64 + w, m = h*2048 + w*32 + ci.
// ---------------------------------------------------------------------------
__global__ void final_kernel(const float* __restrict__ mlp,
                             const float* __restrict__ conv_w,
                             const float* __restrict__ conv_b,
                             float* __restrict__ y)
{
    int p = blockIdx.x * blockDim.x + threadIdx.x;
    if (p >= RH * RW) return;
    float acc = conv_b[0];
    #pragma unroll 4
    for (int c = 0; c < RC; ++c) {
        int n  = p * RC + c;
        int ci = n >> 12;
        int h  = (n >> 6) & 63;
        int w  = n & 63;
        int mrow = h * 2048 + w * 32 + ci;
        acc += conv_w[c] * mlp[mrow];
    }
    y[p] = acc;
}

// ---------------------------------------------------------------------------
extern "C" void kernel_launch(void* const* d_in, const int* in_sizes, int n_in,
                              void* d_out, int out_size, void* d_ws, size_t ws_size,
                              hipStream_t stream) {
    const float* rgb    = (const float*)d_in[0];
    const float* kdtree = (const float*)d_in[1];
    const float* Tr     = (const float*)d_in[2];
    const float* R0     = (const float*)d_in[3];
    const float* P3     = (const float*)d_in[4];
    const float* w1     = (const float*)d_in[5];
    const float* b1     = (const float*)d_in[6];
    const float* w2     = (const float*)d_in[7];
    const float* b2     = (const float*)d_in[8];
    const float* w3     = (const float*)d_in[9];
    const float* b3     = (const float*)d_in[10];
    const float* conv_w = (const float*)d_in[11];
    const float* conv_b = (const float*)d_in[12];

    char* ws = (char*)d_ws;
    _Float16* B1  = (_Float16*)(ws);                 // 139264 B
    _Float16* B2  = (_Float16*)(ws + 139264);        // 16384 B
    float*    M   = (float*)   (ws + 155648);        // 48 B
    float*    mlp = (float*)   (ws + 262144);        // 524288 B
    float*    y   = (float*)d_out;

    int pack_elems = B1_HALVES + B2_HALVES + 12;
    pack_kernel<<<(pack_elems + 255) / 256, 256, 0, stream>>>(w1, w2, Tr, R0, P3, B1, B2, M);
    fused_kernel<<<NROWS / MROWS, 256, 0, stream>>>(rgb, kdtree, M, B1, b1, B2, b2, w3, b3, mlp);
    final_kernel<<<(RH * RW + 255) / 256, 256, 0, stream>>>(mlp, conv_w, conv_b, y);
}
